// Graph_Embedding_57363583205519
// MI455X (gfx1250) — compile-verified, matrix operand order checked by reading, not run on hardware
//
#include <hip/hip_runtime.h>
#include <math.h>

#define NN 2048
#define EE 32768
#define CC 65536
#define ND 256
#define AD 128
#define ED 128
#define KF 640   // 2*ND + ED

typedef unsigned short u16t;
typedef __attribute__((ext_vector_type(16))) __bf16          v16bf;
typedef __attribute__((ext_vector_type(8)))  float           v8f;
typedef __attribute__((ext_vector_type(8)))  unsigned short  us8;

union FragU { v16bf bf; us8 h[2]; };

__device__ __forceinline__ u16t f2bf(float f) {
  union { float f; unsigned u; } v; v.f = f;
  unsigned r = v.u + 0x7FFFu + ((v.u >> 16) & 1u);   // RNE
  return (u16t)(r >> 16);
}

// One 32-wide K chunk: a single A fragment against 4 B tiles -> 4 WMMAs.
__device__ __forceinline__ void kstep(const u16t* __restrict__ pa,
                                      const u16t* const wrow[4], int koff,
                                      int hi, v8f acc_[4]) {
  FragU a;
  a.h[0] = *(const us8*)(pa + hi * 8);
  a.h[1] = *(const us8*)(pa + 16 + hi * 8);
#pragma unroll
  for (int t = 0; t < 4; ++t) {
    FragU b;
    b.h[0] = *(const us8*)(wrow[t] + koff + hi * 8);
    b.h[1] = *(const us8*)(wrow[t] + koff + 16 + hi * 8);
    acc_[t] = __builtin_amdgcn_wmma_f32_16x16x32_bf16(false, a.bf, false, b.bf,
                                                      (short)0, acc_[t], false, false);
  }
}

// ---------------------------------------------------------------------------
// bf16 WMMA GEMM: out = epilogue(A[MxK] @ W[FxK]^T + bias)
// Each wave computes a 16x64 output strip (4 wmma tiles sharing the A frag).
//   GATHER=1 : A rows are per-edge concat(Xh[src], Xh[dst], Xe); K must be 640
//   SCATTER=1: epilogue = relu * scale[row], atomicAdd into out[tgt[row]*F+f]
// Fragments follow the CDNA5 16-bit 16x32 layout: lane l (m=l&15, hi=l>>4)
// holds K = {hi*8..hi*8+7, 16+hi*8..+7} -> two 16-byte loads per lane.
// ---------------------------------------------------------------------------
template<int GATHER, int SCATTER, int K>
__global__ __launch_bounds__(256) void gemm_k(
    const u16t* __restrict__ A,
    const u16t* __restrict__ GXh, const u16t* __restrict__ GXe,
    const int*  __restrict__ gsrc, const int* __restrict__ gdst,
    const u16t* __restrict__ W, const float* __restrict__ bias,
    float* __restrict__ out, int M, int Ft4 /*F/64*/,
    int act, int acc, int outbf,
    const float* __restrict__ scale, const int* __restrict__ tgt)
{
  const int lane = threadIdx.x & 31;
  const int wave = threadIdx.x >> 5;
  const int tile = blockIdx.x * 8 + wave;
  const int mt = tile / Ft4;
  const int f4 = tile - mt * Ft4;
  const int n  = lane & 15;
  const int hi = lane >> 4;
  const int F  = Ft4 * 64;
  const int arow = mt * 16 + n;

  const u16t* wrow[4];
#pragma unroll
  for (int t = 0; t < 4; ++t)
    wrow[t] = W + (size_t)(f4 * 64 + t * 16 + n) * K;

  v8f acc_[4] = {};
  if (!SCATTER && acc) {
#pragma unroll
    for (int t = 0; t < 4; ++t)
#pragma unroll
      for (int v = 0; v < 8; ++v)
        acc_[t][v] = out[(size_t)(mt * 16 + v + 8 * hi) * F + f4 * 64 + t * 16 + n];
  }

  if (GATHER) {
    // hoist the three segment base pointers -> three branch-free K sub-loops
    const u16t* rs = GXh + (size_t)gsrc[arow] * ND;
    const u16t* rd = GXh + (size_t)gdst[arow] * ND;
    const u16t* re = GXe + (size_t)arow * ED;
#pragma unroll 2
    for (int kk = 0; kk < ND; kk += 32) kstep(rs + kk, wrow, kk, hi, acc_);
#pragma unroll 2
    for (int kk = 0; kk < ND; kk += 32) kstep(rd + kk, wrow, ND + kk, hi, acc_);
#pragma unroll 2
    for (int kk = 0; kk < ED; kk += 32) kstep(re + kk, wrow, 2 * ND + kk, hi, acc_);
  } else {
    const u16t* ra = A + (size_t)arow * K;
#pragma unroll 2
    for (int kk = 0; kk < K; kk += 32) kstep(ra + kk, wrow, kk, hi, acc_);
  }

  float bias4[4] = {0.f, 0.f, 0.f, 0.f};
  if (bias) {
#pragma unroll
    for (int t = 0; t < 4; ++t) bias4[t] = bias[f4 * 64 + t * 16 + n];
  }

#pragma unroll
  for (int v = 0; v < 8; ++v) {
    const int m = mt * 16 + v + 8 * hi;
    if (SCATTER) {
      const float sc = scale[m];
      const size_t base = (size_t)tgt[m] * F + f4 * 64 + n;
#pragma unroll
      for (int t = 0; t < 4; ++t) {
        float x = fmaxf(acc_[t][v] + bias4[t], 0.f) * sc;
        atomicAdd(&out[base + t * 16], x);
      }
    } else {
      const size_t base = (size_t)m * F + f4 * 64 + n;
#pragma unroll
      for (int t = 0; t < 4; ++t) {
        float x = acc_[t][v] + bias4[t];
        if (act == 1)      x = fmaxf(x, 0.f);
        else if (act == 2) x = 1.f / (1.f + __expf(-x));
        if (outbf) ((u16t*)out)[base + t * 16] = f2bf(x);
        else       out[base + t * 16] = x;
      }
    }
  }
}

// ------------------------- small helper kernels ----------------------------
__global__ void k_zero(float* p, int n) {
  int i = blockIdx.x * 256 + threadIdx.x;
  if (i < n) p[i] = 0.f;
}
__global__ void k_f2bf(const float* __restrict__ in, u16t* __restrict__ out, int n) {
  int i = blockIdx.x * 256 + threadIdx.x;
  if (i < n) out[i] = f2bf(in[i]);
}
__global__ void k_gather_node(const float* __restrict__ tab, const int* __restrict__ ids,
                              float* __restrict__ XF, u16t* __restrict__ XB) {
  int i = blockIdx.x * 256 + threadIdx.x;
  int r = i >> 8, f = i & 255;
  float v = tab[(size_t)ids[r] * ND + f];
  XF[i] = v; XB[i] = f2bf(v);
}
__global__ void k_gather_bf(const float* __restrict__ tab, const int* __restrict__ ids,
                            u16t* __restrict__ out, int dim, int n) {
  int i = blockIdx.x * 256 + threadIdx.x;
  if (i >= n) return;
  int r = i / dim, f = i - r * dim;
  out[i] = f2bf(tab[(size_t)ids[r] * dim + f]);
}
// ua = Ww1^T @ w2a ; ub = Ww1^T @ w2b   (folds attention scores to rank-1)
__global__ void k_uvec(const float* __restrict__ Ww1, const float* __restrict__ w2a,
                       const float* __restrict__ w2b, float* ua, float* ub) {
  int t = threadIdx.x;
  if (t < AD) {
    float a = 0.f;
    for (int j = 0; j < AD; ++j) a += w2a[j] * Ww1[(size_t)j * AD + t];
    ua[t] = a;
  } else {
    int k = t - AD; float a = 0.f;
    for (int j = 0; j < AD; ++j) a += w2b[j] * Ww1[(size_t)j * AD + k];
    ub[k] = a;
  }
}
__global__ void k_sab(const float* __restrict__ emb_attr, const int* __restrict__ ids,
                      const float* __restrict__ ua, const float* __restrict__ ub,
                      float* sa, float* sb) {
  int r = blockIdx.x * 256 + threadIdx.x;
  if (r >= NN) return;
  const float* x = emb_attr + (size_t)ids[r] * AD;
  float a = 0.f, b = 0.f;
  for (int k = 0; k < AD; ++k) { a += x[k] * ua[k]; b += x[k] * ub[k]; }
  sa[r] = a; sb[r] = b;
}
// per-row masked softmax stats (max + sum) over w[r,c] = sa[r]+sb[c]
__global__ void k_rowstats(const int* __restrict__ adj, const float* __restrict__ sa,
                           const float* __restrict__ sb, float* rmax, float* rsum) {
  int r = blockIdx.x;
  __shared__ float red[256];
  float mx = -3.0e38f;
  for (int c = threadIdx.x; c < NN; c += 256)
    if (adj[(size_t)r * NN + c] > 0) mx = fmaxf(mx, sa[r] + sb[c]);
  red[threadIdx.x] = mx; __syncthreads();
  for (int s = 128; s > 0; s >>= 1) {
    if (threadIdx.x < s) red[threadIdx.x] = fmaxf(red[threadIdx.x], red[threadIdx.x + s]);
    __syncthreads();
  }
  mx = red[0]; __syncthreads();
  float sm = 0.f;
  for (int c = threadIdx.x; c < NN; c += 256)
    if (adj[(size_t)r * NN + c] > 0) sm += __expf(sa[r] + sb[c] - mx);
  red[threadIdx.x] = sm; __syncthreads();
  for (int s = 128; s > 0; s >>= 1) {
    if (threadIdx.x < s) red[threadIdx.x] += red[threadIdx.x + s];
    __syncthreads();
  }
  if (threadIdx.x == 0) { rmax[r] = mx; rsum[r] = red[0]; }
}
__global__ void k_edge_att(const int* __restrict__ sI, const int* __restrict__ dI,
                           const float* __restrict__ sa, const float* __restrict__ sb,
                           const float* __restrict__ rmax, const float* __restrict__ rsum,
                           float* ap, float* ac) {
  int e = blockIdx.x * 256 + threadIdx.x;
  if (e >= EE) return;
  int s = sI[e], d = dI[e];
  ap[e] = __expf(sa[d] + sb[s] - rmax[d]) / rsum[d];   // att[dst,src]
  ac[e] = __expf(sa[s] + sb[d] - rmax[s]) / rsum[s];   // att[src,dst]
}
__global__ void k_norms(const float* __restrict__ X, float* nrm) {
  int r = blockIdx.x;
  __shared__ float red[256];
  float v = X[(size_t)r * ND + threadIdx.x];
  red[threadIdx.x] = v * v; __syncthreads();
  for (int s = 128; s > 0; s >>= 1) {
    if (threadIdx.x < s) red[threadIdx.x] += red[threadIdx.x + s];
    __syncthreads();
  }
  if (threadIdx.x == 0) nrm[r] = fmaxf(sqrtf(red[0]), 1e-8f);
}
// pass1: ex[c] = exp(cos_sim), den[i] += ex  (sim in [-1,1] -> shift unneeded)
__global__ void k_cross1(const float* __restrict__ X1, const float* __restrict__ X2,
                         const int* __restrict__ iA, const int* __restrict__ jA,
                         const float* __restrict__ n1, const float* __restrict__ n2,
                         float* ex, float* den) {
  int c = blockIdx.x * 256 + threadIdx.x;
  if (c >= CC) return;
  int i = iA[c], j = jA[c];
  const float4* x1 = (const float4*)(X1 + (size_t)i * ND);
  const float4* x2 = (const float4*)(X2 + (size_t)j * ND);
  float dp = 0.f;
  for (int f = 0; f < ND / 4; ++f) {
    float4 a = x1[f], b = x2[f];
    dp += a.x * b.x + a.y * b.y + a.z * b.z + a.w * b.w;
  }
  float e = __expf(dp / (n1[i] * n2[j]));
  ex[c] = e;
  atomicAdd(&den[i], e);
}
// pass2: Agg[i] += (ex/den[i]) * X2[j]   (one wave per pair)
__global__ void k_cross2(const float* __restrict__ X2,
                         const int* __restrict__ iA, const int* __restrict__ jA,
                         const float* __restrict__ ex, const float* __restrict__ den,
                         float* Agg) {
  int c = blockIdx.x * 8 + (threadIdx.x >> 5);
  int lane = threadIdx.x & 31;
  if (c >= CC) return;
  int i = iA[c], j = jA[c];
  float w = ex[c] / den[i];
  for (int f = lane; f < ND; f += 32)
    atomicAdd(&Agg[(size_t)i * ND + f], w * X2[(size_t)j * ND + f]);
}
__global__ void k_xc(const float* __restrict__ Xm, const float* __restrict__ Gate,
                     const float* __restrict__ Agg, u16t* XcB, int n) {
  int i = blockIdx.x * 256 + threadIdx.x;
  if (i < n) XcB[i] = f2bf(Xm[i] + Gate[i] * Agg[i]);
}
__global__ void k_gru(const float* __restrict__ Lr, const float* __restrict__ Lu,
                      const float* __restrict__ Lh1, const float* __restrict__ Lh2,
                      float* XhF, u16t* XhB, int n) {
  int i = blockIdx.x * 256 + threadIdx.x;
  if (i >= n) return;
  float r = 1.f / (1.f + __expf(-Lr[i]));
  float u = 1.f / (1.f + __expf(-Lu[i]));
  float h = tanhf(Lh1[i] + r * Lh2[i]);
  float y = (1.f - u) * XhF[i] + u * h;
  XhF[i] = y; XhB[i] = f2bf(y);
}
__global__ void k_gprod(const float* __restrict__ G1, const float* __restrict__ G2, float* s) {
  int f = blockIdx.x;
  __shared__ float red[256];
  float acc = 0.f;
  for (int r = threadIdx.x; r < NN; r += 256)
    acc += G1[(size_t)r * ND + f] * G2[(size_t)r * ND + f];
  red[threadIdx.x] = acc; __syncthreads();
  for (int t = 128; t > 0; t >>= 1) {
    if (threadIdx.x < t) red[threadIdx.x] += red[threadIdx.x + t];
    __syncthreads();
  }
  if (threadIdx.x == 0) s[f] = red[0];
}
__global__ void k_final(const float* __restrict__ W, const float* __restrict__ b,
                        const float* __restrict__ s, float* out) {
  int n = threadIdx.x;
  float acc = b[n];
  for (int k = 0; k < ND; ++k) acc += W[(size_t)n * ND + k] * s[k];
  out[n] = fmaxf(acc, 0.f);
}

// ---------------------------------------------------------------------------
extern "C" void kernel_launch(void* const* d_in, const int* in_sizes, int n_in,
                              void* d_out, int out_size, void* d_ws, size_t ws_size,
                              hipStream_t stream) {
  (void)in_sizes; (void)n_in; (void)out_size; (void)ws_size;
  const int* node_ids[2]  = {(const int*)d_in[0],  (const int*)d_in[1]};
  const int* attr_ids[2]  = {(const int*)d_in[2],  (const int*)d_in[3]};
  const int* edge_attr[2] = {(const int*)d_in[4],  (const int*)d_in[5]};
  const int* indices[2]   = {(const int*)d_in[6],  (const int*)d_in[7]};
  const int* adj[2]       = {(const int*)d_in[8],  (const int*)d_in[9]};
  const int* cross[2]     = {(const int*)d_in[10], (const int*)d_in[11]};
  const float* emb_node = (const float*)d_in[12];
  const float* emb_edge = (const float*)d_in[13];
  const float* emb_attr = (const float*)d_in[14];
  const float* Ww1 = (const float*)d_in[15];
  const float* w2a = (const float*)d_in[16];
  const float* w2b = (const float*)d_in[17];
  const float* Wp1 = (const float*)d_in[18]; const float* bp1 = (const float*)d_in[19];
  const float* Wp2 = (const float*)d_in[20]; const float* bp2 = (const float*)d_in[21];
  const float* Wc1 = (const float*)d_in[22]; const float* bc1 = (const float*)d_in[23];
  const float* Wc2 = (const float*)d_in[24]; const float* bc2 = (const float*)d_in[25];
  const float* Wgate = (const float*)d_in[26];
  const float* Wrm = (const float*)d_in[27]; const float* brm = (const float*)d_in[28];
  const float* Wrh = (const float*)d_in[29]; const float* brh = (const float*)d_in[30];
  const float* Wum = (const float*)d_in[31]; const float* bum = (const float*)d_in[32];
  const float* Wuh = (const float*)d_in[33]; const float* buh = (const float*)d_in[34];
  const float* Whm = (const float*)d_in[35]; const float* bhm = (const float*)d_in[36];
  const float* Whh = (const float*)d_in[37]; const float* bhh = (const float*)d_in[38];
  const float* Wfn = (const float*)d_in[39]; const float* bfn = (const float*)d_in[40];
  const float* Wfg = (const float*)d_in[41]; const float* bfg = (const float*)d_in[42];
  const float* WfG = (const float*)d_in[43]; const float* bfG = (const float*)d_in[44];

  // ---- workspace bump allocator (256B aligned) ----
  size_t off = 0;
  auto alloc = [&](size_t bytes) -> void* {
    off = (off + 255) & ~(size_t)255;
    void* p = (char*)d_ws + off;
    off += bytes;
    return p;
  };
  float* XhF[2]; u16t* XhB[2]; u16t* XnB[2]; u16t* XeB[2];
  for (int g = 0; g < 2; ++g) {
    XhF[g] = (float*)alloc((size_t)NN * ND * 4);
    XhB[g] = (u16t*)alloc((size_t)NN * ND * 2);
    XnB[g] = (u16t*)alloc((size_t)NN * AD * 2);
    XeB[g] = (u16t*)alloc((size_t)EE * ED * 2);
  }
  u16t* Wp1B = (u16t*)alloc((size_t)ND * KF * 2);
  u16t* Wc1B = (u16t*)alloc((size_t)ND * KF * 2);
  u16t *Wp2B=(u16t*)alloc((size_t)ND*ND*2), *Wc2B=(u16t*)alloc((size_t)ND*ND*2);
  u16t *WrmB=(u16t*)alloc((size_t)ND*ND*2), *WrhB=(u16t*)alloc((size_t)ND*ND*2);
  u16t *WumB=(u16t*)alloc((size_t)ND*ND*2), *WuhB=(u16t*)alloc((size_t)ND*ND*2);
  u16t *WhmB=(u16t*)alloc((size_t)ND*ND*2), *WhhB=(u16t*)alloc((size_t)ND*ND*2);
  u16t *WfnB=(u16t*)alloc((size_t)ND*ND*2);
  u16t *WgateB=(u16t*)alloc((size_t)ND*AD*2), *WfgB=(u16t*)alloc((size_t)ND*AD*2);
  float *ua=(float*)alloc(AD*4), *ub=(float*)alloc(AD*4);
  float *sa[2], *sb[2], *rmax[2], *rsum[2], *ap[2], *ac[2];
  float *Xm[2], *Gate[2], *Agg[2], *den[2], *nrm[2]; u16t* XcB[2];
  for (int g = 0; g < 2; ++g) {
    sa[g]=(float*)alloc(NN*4);   sb[g]=(float*)alloc(NN*4);
    rmax[g]=(float*)alloc(NN*4); rsum[g]=(float*)alloc(NN*4);
    ap[g]=(float*)alloc(EE*4);   ac[g]=(float*)alloc(EE*4);
    Xm[g]=(float*)alloc((size_t)NN*ND*4);
    Gate[g]=(float*)alloc((size_t)NN*ND*4);
    Agg[g]=(float*)alloc((size_t)NN*ND*4);
    den[g]=(float*)alloc(NN*4);  nrm[g]=(float*)alloc(NN*4);
    XcB[g]=(u16t*)alloc((size_t)NN*ND*2);
  }
  u16t* H1B = (u16t*)alloc((size_t)EE * ND * 2);   // shared layer-1 activations
  float* ex = (float*)alloc((size_t)CC * 4);
  float* Lr = (float*)alloc((size_t)NN*ND*4);
  float* Lu = (float*)alloc((size_t)NN*ND*4);
  float* Lh1= (float*)alloc((size_t)NN*ND*4);
  float* Lh2= (float*)alloc((size_t)NN*ND*4);
  float* svec = (float*)alloc(ND*4);

  auto zg = [](int n) { return dim3((n + 255) / 256); };
  auto gemm = [&](int gather, int scatter, const u16t* A, const u16t* GXh,
                  const u16t* GXe, const int* sI, const int* dI,
                  const u16t* W, const float* bias, void* out,
                  int M, int K, int F, int act, int accf, int outbf,
                  const float* scale, const int* tgt) {
    int blocks = (M / 16) * (F / 64) / 8;
    if (gather)
      gemm_k<1,0,KF><<<blocks,256,0,stream>>>(A,GXh,GXe,sI,dI,W,bias,(float*)out,M,F/64,act,accf,outbf,scale,tgt);
    else if (scatter)
      gemm_k<0,1,ND><<<blocks,256,0,stream>>>(A,GXh,GXe,sI,dI,W,bias,(float*)out,M,F/64,act,accf,outbf,scale,tgt);
    else if (K == ND)
      gemm_k<0,0,ND><<<blocks,256,0,stream>>>(A,GXh,GXe,sI,dI,W,bias,(float*)out,M,F/64,act,accf,outbf,scale,tgt);
    else
      gemm_k<0,0,AD><<<blocks,256,0,stream>>>(A,GXh,GXe,sI,dI,W,bias,(float*)out,M,F/64,act,accf,outbf,scale,tgt);
  };

  // ---------------- setup: bf16 weights, embeddings, attention ----------------
  k_f2bf<<<zg(ND*KF),256,0,stream>>>(Wp1, Wp1B, ND*KF);
  k_f2bf<<<zg(ND*KF),256,0,stream>>>(Wc1, Wc1B, ND*KF);
  k_f2bf<<<zg(ND*ND),256,0,stream>>>(Wp2, Wp2B, ND*ND);
  k_f2bf<<<zg(ND*ND),256,0,stream>>>(Wc2, Wc2B, ND*ND);
  k_f2bf<<<zg(ND*ND),256,0,stream>>>(Wrm, WrmB, ND*ND);
  k_f2bf<<<zg(ND*ND),256,0,stream>>>(Wrh, WrhB, ND*ND);
  k_f2bf<<<zg(ND*ND),256,0,stream>>>(Wum, WumB, ND*ND);
  k_f2bf<<<zg(ND*ND),256,0,stream>>>(Wuh, WuhB, ND*ND);
  k_f2bf<<<zg(ND*ND),256,0,stream>>>(Whm, WhmB, ND*ND);
  k_f2bf<<<zg(ND*ND),256,0,stream>>>(Whh, WhhB, ND*ND);
  k_f2bf<<<zg(ND*ND),256,0,stream>>>(Wfn, WfnB, ND*ND);
  k_f2bf<<<zg(ND*AD),256,0,stream>>>(Wgate, WgateB, ND*AD);
  k_f2bf<<<zg(ND*AD),256,0,stream>>>(Wfg, WfgB, ND*AD);
  k_uvec<<<1,256,0,stream>>>(Ww1, w2a, w2b, ua, ub);
  for (int g = 0; g < 2; ++g) {
    k_gather_node<<<zg(NN*ND),256,0,stream>>>(emb_node, node_ids[g], XhF[g], XhB[g]);
    k_gather_bf<<<zg(NN*AD),256,0,stream>>>(emb_attr, attr_ids[g], XnB[g], AD, NN*AD);
    k_gather_bf<<<zg(EE*ED),256,0,stream>>>(emb_edge, edge_attr[g], XeB[g], ED, EE*ED);
    k_sab<<<zg(NN),256,0,stream>>>(emb_attr, attr_ids[g], ua, ub, sa[g], sb[g]);
    k_rowstats<<<NN,256,0,stream>>>(adj[g], sa[g], sb[g], rmax[g], rsum[g]);
    k_edge_att<<<zg(EE),256,0,stream>>>(indices[g], indices[g]+EE, sa[g], sb[g],
                                        rmax[g], rsum[g], ap[g], ac[g]);
    // gate = sigmoid(Xn @ Wgate^T) -- loop-invariant
    gemm(0,0, XnB[g],0,0,0,0, WgateB, nullptr, Gate[g], NN, AD, ND, 2, 0, 0, 0, 0);
  }

  // ------------------------------ prop steps --------------------------------
  for (int step = 0; step < 3; ++step) {
    for (int g = 0; g < 2; ++g) {
      const int* sI = indices[g]; const int* dI = indices[g] + EE;
      k_zero<<<zg(NN*ND),256,0,stream>>>(Xm[g], NN*ND);
      // prop branch: H1 = relu(feat@Wp1^T+bp1); Xm[dst] += ap * relu(H1@Wp2^T+bp2)
      gemm(1,0, nullptr, XhB[g], XeB[g], sI, dI, Wp1B, bp1, H1B, EE, KF, ND, 1,0,1, 0,0);
      gemm(0,1, H1B, 0,0,0,0, Wp2B, bp2, Xm[g], EE, ND, ND, 1,0,0, ap[g], dI);
      // comm branch: scatter to src with ac
      gemm(1,0, nullptr, XhB[g], XeB[g], sI, dI, Wc1B, bc1, H1B, EE, KF, ND, 1,0,1, 0,0);
      gemm(0,1, H1B, 0,0,0,0, Wc2B, bc2, Xm[g], EE, ND, ND, 1,0,0, ac[g], sI);
    }
    // cross-graph attention (uses pre-update Xh of both graphs)
    for (int g = 0; g < 2; ++g) {
      k_zero<<<zg(NN),256,0,stream>>>(den[g], NN);
      k_zero<<<zg(NN*ND),256,0,stream>>>(Agg[g], NN*ND);
      k_norms<<<NN,256,0,stream>>>(XhF[g], nrm[g]);
    }
    for (int g = 0; g < 2; ++g) {
      int o = g ^ 1;
      const int* iA = cross[g]; const int* jA = cross[g] + CC;
      k_cross1<<<zg(CC),256,0,stream>>>(XhF[g], XhF[o], iA, jA, nrm[g], nrm[o], ex, den[g]);
      k_cross2<<<CC/8,256,0,stream>>>(XhF[o], iA, jA, ex, den[g], Agg[g]);
    }
    for (int g = 0; g < 2; ++g)
      k_xc<<<zg(NN*ND),256,0,stream>>>(Xm[g], Gate[g], Agg[g], XcB[g], NN*ND);
    // GRU update (per graph: 6 WMMA GEMMs + elementwise)
    for (int g = 0; g < 2; ++g) {
      gemm(0,0, XcB[g],0,0,0,0, WrmB, brm, Lr,  NN, ND, ND, 0,0,0, 0,0);
      gemm(0,0, XhB[g],0,0,0,0, WrhB, brh, Lr,  NN, ND, ND, 0,1,0, 0,0);
      gemm(0,0, XcB[g],0,0,0,0, WumB, bum, Lu,  NN, ND, ND, 0,0,0, 0,0);
      gemm(0,0, XhB[g],0,0,0,0, WuhB, buh, Lu,  NN, ND, ND, 0,1,0, 0,0);
      gemm(0,0, XcB[g],0,0,0,0, WhmB, bhm, Lh1, NN, ND, ND, 0,0,0, 0,0);
      gemm(0,0, XhB[g],0,0,0,0, WhhB, bhh, Lh2, NN, ND, ND, 0,0,0, 0,0);
      k_gru<<<zg(NN*ND),256,0,stream>>>(Lr, Lu, Lh1, Lh2, XhF[g], XhB[g], NN*ND);
    }
  }

  // ------------------------------ aggregate ---------------------------------
  for (int g = 0; g < 2; ++g) {
    gemm(0,0, XnB[g],0,0,0,0, WfgB, bfg, Lr, NN, AD, ND, 2,0,0, 0,0);  // sigmoid gate
    gemm(0,0, XhB[g],0,0,0,0, WfnB, bfn, Lu, NN, ND, ND, 0,0,0, 0,0);  // linear
    k_gprod<<<ND,256,0,stream>>>(Lr, Lu, svec);
    k_final<<<1,256,0,stream>>>(WfG, bfG, svec, (float*)d_out + g * ND);
  }
}
